// Transformer_encoder_45749991637535
// MI455X (gfx1250) — matrix operand order repaired and checked by reading, not compiled, not run on hardware
//
#include <hip/hip_runtime.h>

// ---------------------------------------------------------------------------
// B=2, C=64, H=W=D=64.  x: [B,C,H,W,D] f32 (sC=64^3, sH=4096, sW=64, sD=1).
// Strategy:
//   * pre-pack weights to bf16 in WMMA-fragment order (1 contiguous 32B load/frag)
//   * pre-transpose x into padded [B][66][66][66][64] bf16 (channel-innermost)
//     -> conv input staging is a pure 3-D slab copy: done by the TDM
//     -> B-fragments are single aligned 32B LDS vector loads
//   * conv epilogue emits bf16 q/k/v (q pre-scaled by 1/8) for the 6 axial passes
//   * axial passes are templated on the axis permutation: constant strides +
//     vectorized staging on whichever of L/M is the contiguous D axis
//   * 2nd axial pass of each pair writes bf16 into padded cat-tensor AVt[...][192]
//   * final conv stages AVt slabs via TDM, fused bias+BN*ReLU*BN + residual(+q)
// ---------------------------------------------------------------------------
typedef __attribute__((ext_vector_type(16))) __bf16 v16bf;
typedef __attribute__((ext_vector_type(8)))  __bf16 v8bf;
typedef __attribute__((ext_vector_type(8)))  float  v8f;
typedef __attribute__((ext_vector_type(4)))  unsigned int u32x4;
typedef __attribute__((ext_vector_type(4)))  int i32x4;
typedef __attribute__((ext_vector_type(8)))  int i32x8;

#define BN_SCALE 0.9999950000374997f   // 1/sqrt(1+1e-5)
#define NEL      33554432L             // 2*64^4 elements per BCHWD tensor

#if defined(__HIP_DEVICE_COMPILE__) && __has_builtin(__builtin_amdgcn_tensor_load_to_lds)
#define HAVE_TDM 1
#else
#define HAVE_TDM 0
#endif

union BF16x16 { v16bf v; v8bf h[2]; };

// WMMA 16-bit A-fragment K pattern: one contiguous 16-run per lane half (matches B).
__device__ __forceinline__ int a_kidx(int e, int hf) {
    int base = hf * 8;
    return 2 * base + e;
}

#if HAVE_TDM
// Build a D# for a 3-D tile (dim0 contiguous, dims 1/2 strided) and issue
// TENSOR_LOAD_TO_LDS (CDNA5 ISA).  data_size=1 (2 bytes).
__device__ __forceinline__ void tdm_load(
    unsigned lds_off, const void* gaddr,
    unsigned tile0, unsigned tile1, unsigned tile2,
    unsigned long long str01, unsigned long long str12,  // tensor_dim0/1_stride
    unsigned td0, unsigned td1, unsigned td2)
{
    unsigned long long ga = (unsigned long long)gaddr;
    u32x4 g0;
    g0[0] = 1u;                                           // count=1, user mode
    g0[1] = lds_off;                                      // lds_addr
    g0[2] = (unsigned)(ga & 0xFFFFFFFFu);                 // global_addr lo
    g0[3] = (unsigned)((ga >> 32) & 0x1FFFFFFu) | (2u << 30);  // addr hi | type=2
    unsigned long long q0 = (1ull << 16)                                  // data_size=2B
                          | ((unsigned long long)(td0 & 0xFFFFu) << 48);  // tdim0[15:0]
    unsigned long long q1 = ((unsigned long long)(td0 >> 16) & 0xFFFFull)
                          | ((unsigned long long)td1 << 16)               // tdim1
                          | ((unsigned long long)(tile0 & 0xFFFFu) << 48);
    unsigned long long q2 = (unsigned long long)(tile1 & 0xFFFFu)
                          | ((unsigned long long)(tile2 & 0xFFFFu) << 16)
                          | ((str01 & 0xFFFFFFFFull) << 32);
    unsigned long long q3 = ((str01 >> 32) & 0xFFFFull)
                          | ((str12 & 0xFFFFFFFFFFFFull) << 16);
    i32x8 g1;
    g1[0] = (int)(unsigned)q0; g1[1] = (int)(unsigned)(q0 >> 32);
    g1[2] = (int)(unsigned)q1; g1[3] = (int)(unsigned)(q1 >> 32);
    g1[4] = (int)(unsigned)q2; g1[5] = (int)(unsigned)(q2 >> 32);
    g1[6] = (int)(unsigned)q3; g1[7] = (int)(unsigned)(q3 >> 32);
    i32x4 g2; g2[0] = (int)td2; g2[1] = 0; g2[2] = 0; g2[3] = 0;
    i32x4 g3; g3[0] = 0; g3[1] = 0; g3[2] = 0; g3[3] = 0;
#if defined(__clang_major__) && __clang_major__ >= 23
    i32x8 g4 = {};
    __builtin_amdgcn_tensor_load_to_lds(g0, g1, g2, g3, g4, 0);
#else
    __builtin_amdgcn_tensor_load_to_lds(g0, g1, g2, g3, 0);
#endif
}
#endif  // HAVE_TDM

// ===========================================================================
// Prep kernels
// ===========================================================================
__global__ void __launch_bounds__(256) zero_pad_kernel(u32x4* dst, long nchunks) {
    long i = (long)blockIdx.x * 256 + threadIdx.x;
    u32x4 z = {};
    if (i < nchunks) dst[i] = z;
}

// x [B,C,H,W,D] f32 -> padded transposed xt [B][66][66][66][64] bf16
__global__ void __launch_bounds__(256) xpack_kernel(
    const float* __restrict__ x, __bf16* __restrict__ xt) {
    long i = (long)blockIdx.x * 256 + threadIdx.x;
    if (i >= NEL) return;
    int d = (int)(i & 63), w = (int)((i >> 6) & 63), h = (int)((i >> 12) & 63);
    int c = (int)((i >> 18) & 63), b = (int)(i >> 24);
    xt[((((long)b * 66 + h + 1) * 66 + (w + 1)) * 66 + (d + 1)) * 64 + c] = (__bf16)x[i];
}

// weights [Cout=64][Cin][27] f32 -> bf16 in WMMA-fragment order
// [tap][kc][mi][lane][e], kc counts 32-wide cin chunks (kcs = Cin/32).
__global__ void __launch_bounds__(256) wpack_kernel(
    const float* __restrict__ wsrc, __bf16* __restrict__ wdst, int Cin, int kcs) {
    long total = 27L * kcs * 2048;
    long i = (long)blockIdx.x * 256 + threadIdx.x;
    if (i >= total) return;
    int e    = (int)(i & 15);
    int lane = (int)((i >> 4) & 31);
    int mi   = (int)((i >> 9) & 3);
    long r   = i >> 11;
    int kc   = (int)(r % kcs);
    int tap  = (int)(r / kcs);
    int hf = lane >> 4, l15 = lane & 15;
    int co = 16 * mi + l15;
    int ci = 32 * kc + a_kidx(e, hf);
    wdst[i] = (__bf16)wsrc[(co * Cin + ci) * 27 + tap];
}

// ===========================================================================
// Kernel 1: fused q/k/v implicit-GEMM conv.  M=64 couts, N=32 d-strip,
// K=64cin*27taps.  x tile staged by the TDM; 162 wmma per wave.
// ===========================================================================
__global__ void __launch_bounds__(256) qkv_conv_kernel(
    const __bf16* __restrict__ xt,
    const __bf16* __restrict__ wqp, const __bf16* __restrict__ wkp,
    const __bf16* __restrict__ wvp,
    const float* __restrict__ bq, const float* __restrict__ bk,
    const float* __restrict__ bv,
    float* __restrict__ qo, __bf16* __restrict__ qh,
    __bf16* __restrict__ kh, __bf16* __restrict__ vh)
{
    __shared__ __attribute__((aligned(32))) __bf16 sx[3][3][34][64];  // [dh][dw][d][c] 38KB

    const int tid = threadIdx.x, blk = blockIdx.x;
    const int dt = blk & 1, w = (blk >> 1) & 63, h = (blk >> 7) & 63, b = blk >> 13;
    const int d0 = dt * 32;

    const long origin = ((((long)b * 66 + h) * 66 + w) * 66 + d0) * 64;  // (h-1,w-1,d0-1) padded
#if HAVE_TDM
    if (tid == 0) {
        tdm_load((unsigned)(unsigned long long)(&sx[0][0][0][0]), xt + origin,
                 34 * 64, 3, 3, 66 * 64, 66 * 66 * 64, 34 * 64, 3, 3);
        __builtin_amdgcn_s_wait_tensorcnt(0);
    }
#else
    for (int ch = tid; ch < 2448; ch += 256) {           // 9 slabs * 272 x 16B
        int s9 = ch / 272, rem = ch % 272;
        int dh = s9 / 3, dw = s9 % 3;
        const u32x4* src =
            (const u32x4*)(xt + origin + (long)dh * (66 * 66 * 64) + (long)dw * (66 * 64));
        *((u32x4*)&sx[dh][dw][0][0] + rem) = src[rem];
    }
#endif
    __syncthreads();

    const int lane = tid & 31, wave = tid >> 5;
    const int mi = wave >> 1, ni = wave & 1;
    const int hf = lane >> 4, l15 = lane & 15;
    const int col = 16 * ni + l15;
    const long fbase = (long)mi * 512 + lane * 16;

    v8f accQ = {}; v8f accK = {}; v8f accV = {};

    for (int tap = 0; tap < 27; ++tap) {
        const int dh = tap / 9, dw = (tap / 3) % 3, dd = tap % 3;
        for (int kc = 0; kc < 2; ++kc) {
            const long fo = fbase + (long)(tap * 2 + kc) * 2048;
            v16bf aQ = *(const v16bf*)(wqp + fo);
            v16bf aK = *(const v16bf*)(wkp + fo);
            v16bf aV = *(const v16bf*)(wvp + fo);
            v16bf bX = *(const v16bf*)&sx[dh][dw][col + dd][32 * kc + 16 * hf];
            accQ = __builtin_amdgcn_wmma_f32_16x16x32_bf16(false, aQ, false, bX,
                                                           (short)0, accQ, false, false);
            accK = __builtin_amdgcn_wmma_f32_16x16x32_bf16(false, aK, false, bX,
                                                           (short)0, accK, false, false);
            accV = __builtin_amdgcn_wmma_f32_16x16x32_bf16(false, aV, false, bX,
                                                           (short)0, accV, false, false);
        }
    }

#pragma unroll
    for (int r = 0; r < 8; ++r) {
        int  co   = 16 * mi + r + 8 * hf;
        int  d    = d0 + 16 * ni + l15;
        long oidx = ((((long)b * 64 + co) * 64 + h) * 64 + w) * 64 + d;
        float qv = accQ[r] + bq[co];
        qo[oidx] = qv;                        // f32 q for the final residual
        qh[oidx] = (__bf16)(qv * 0.125f);     // 1/DK folded into Q
        kh[oidx] = (__bf16)(accK[r] + bk[co]);
        vh[oidx] = (__bf16)(accV[r] + bv[co]);
    }
}

// ===========================================================================
// Kernel 2: one axial pass, templated on the (S,L,M)->(H,W,D) axis codes so
// strides are compile-time and staging vectorizes on the contiguous axis.
// scores=(Q/8)K^T, softmax over i (axis=-2), out=A*V.
// INIT pass writes f32 accumulator; final pass writes bf16 into padded AVt.
// ===========================================================================
template <int AS, int AL, int AM, int DST, int INIT>
__global__ void __launch_bounds__(256) axial_kernel(
    const __bf16* __restrict__ qh, const __bf16* __restrict__ kh,
    const __bf16* __restrict__ vh,
    float* __restrict__ avf, __bf16* __restrict__ avt)
{
    __shared__ __attribute__((aligned(32))) __bf16 Qs[64][64];   // Q, then softmaxed A
    __shared__ __attribute__((aligned(32))) __bf16 Ks[64][64];
    __shared__ __attribute__((aligned(32))) __bf16 VsT[64][64];  // V transposed [m][j]
    __shared__ float Sc[64][64];
    __shared__ float pmax[4][64];
    __shared__ float psum[4][64];

    constexpr int strS = AS == 0 ? 4096 : (AS == 1 ? 64 : 1);
    constexpr int strL = AL == 0 ? 4096 : (AL == 1 ? 64 : 1);
    constexpr int strM = AM == 0 ? 4096 : (AM == 1 ? 64 : 1);

    const int tid = threadIdx.x, blk = blockIdx.x;
    const int s = blk & 63, c = (blk >> 6) & 63, b = blk >> 12;
    const long base = ((long)(b * 64 + c) << 18) + (long)s * strS;

    if (tid == 0 && s + 1 < 64)     // hint next slice toward GL2
        __builtin_prefetch(qh + base + strS, 0, 1);

    if (AM == 2) {
        // M axis contiguous: vector loads + vector stores for Q/K, scatter for V^T
        for (int ch = tid; ch < 512; ch += 256) {
            int  l = ch >> 3, m0 = (ch & 7) * 8;
            long g = base + (long)l * strL + m0;
            v8bf q8 = *(const v8bf*)(qh + g);
            v8bf k8 = *(const v8bf*)(kh + g);
            v8bf v8 = *(const v8bf*)(vh + g);
            *(v8bf*)&Qs[l][m0] = q8;
            *(v8bf*)&Ks[l][m0] = k8;
#pragma unroll
            for (int e = 0; e < 8; ++e) VsT[m0 + e][l] = v8[e];
        }
    } else if (AL == 2) {
        // L axis contiguous: vector loads + vector stores for V^T, scatter Q/K
        for (int ch = tid; ch < 512; ch += 256) {
            int  m = ch >> 3, l0 = (ch & 7) * 8;
            long g = base + (long)m * strM + l0;
            v8bf q8 = *(const v8bf*)(qh + g);
            v8bf k8 = *(const v8bf*)(kh + g);
            v8bf v8 = *(const v8bf*)(vh + g);
            *(v8bf*)&VsT[m][l0] = v8;
#pragma unroll
            for (int e = 0; e < 8; ++e) { Qs[l0 + e][m] = q8[e]; Ks[l0 + e][m] = k8[e]; }
        }
    } else {
        // S axis contiguous (pa/pb): scalar gather
        for (int idx = tid; idx < 4096; idx += 256) {
            int  l = idx >> 6, m = idx & 63;
            long g = base + (long)l * strL + (long)m * strM;
            Qs[l][m]  = qh[g];
            Ks[l][m]  = kh[g];
            VsT[m][l] = vh[g];
        }
    }
    __syncthreads();

    const int lane = tid & 31, wave = tid >> 5;
    const int hf = lane >> 4, l15 = lane & 15;
    const int it = wave >> 1;

    // ---- GEMM1: Sc[i][j] = sum_m Qs[i][m]*Ks[j][m] ----
    for (int jn = 0; jn < 2; ++jn) {
        const int jt = (wave & 1) * 2 + jn;
        v8f acc = {};
#pragma unroll
        for (int kc = 0; kc < 2; ++kc) {
            BF16x16 a;
            a.h[0] = *(const v8bf*)&Qs[16 * it + l15][32 * kc + 16 * hf];
            a.h[1] = *(const v8bf*)&Qs[16 * it + l15][32 * kc + 16 * hf + 8];
            v16bf bF = *(const v16bf*)&Ks[16 * jt + l15][32 * kc + 16 * hf];
            acc = __builtin_amdgcn_wmma_f32_16x16x32_bf16(false, a.v, false, bF,
                                                          (short)0, acc, false, false);
        }
#pragma unroll
        for (int r = 0; r < 8; ++r)
            Sc[16 * it + r + 8 * hf][16 * jt + l15] = acc[r];
    }
    __syncthreads();

    // ---- softmax over i (axis=-2), 4 row-chunks per column ----
    const int p = tid >> 6, j = tid & 63;
    {
        float mx = -3.402823466e38f;
        for (int i = 16 * p; i < 16 * p + 16; ++i) mx = fmaxf(mx, Sc[i][j]);
        pmax[p][j] = mx;
    }
    __syncthreads();
    const float mx = fmaxf(fmaxf(pmax[0][j], pmax[1][j]), fmaxf(pmax[2][j], pmax[3][j]));
    {
        float sum = 0.0f;
        for (int i = 16 * p; i < 16 * p + 16; ++i) {
            float e = __expf(Sc[i][j] - mx);
            Sc[i][j] = e;
            sum += e;
        }
        psum[p][j] = sum;
    }
    __syncthreads();
    const float rs = 1.0f / (psum[0][j] + psum[1][j] + psum[2][j] + psum[3][j]);
    for (int i = 16 * p; i < 16 * p + 16; ++i)
        Qs[i][j] = (__bf16)(Sc[i][j] * rs);
    __syncthreads();

    // ---- GEMM2: out[i][m] = sum_j A[i][j]*V[j][m] ----
    for (int mn = 0; mn < 2; ++mn) {
        const int mt = (wave & 1) * 2 + mn;
        v8f acc = {};
#pragma unroll
        for (int kc = 0; kc < 2; ++kc) {
            BF16x16 a;
            a.h[0] = *(const v8bf*)&Qs[16 * it + l15][32 * kc + 16 * hf];
            a.h[1] = *(const v8bf*)&Qs[16 * it + l15][32 * kc + 16 * hf + 8];
            v16bf bF = *(const v16bf*)&VsT[16 * mt + l15][32 * kc + 16 * hf];
            acc = __builtin_amdgcn_wmma_f32_16x16x32_bf16(false, a.v, false, bF,
                                                          (short)0, acc, false, false);
        }
#pragma unroll
        for (int r = 0; r < 8; ++r) {
            int   i   = 16 * it + r + 8 * hf;
            int   m   = 16 * mt + l15;
            long  g   = base + (long)i * strL + (long)m * strM;
            float val = 0.5f * acc[r];
            if (INIT) {
                avf[g] = val;                         // first pass of the pair
            } else {
                val += avf[g];                        // pair-sum; emit bf16 cat slot
                int hh = AS == 0 ? s : (AL == 0 ? i : m);
                int ww = AS == 1 ? s : (AL == 1 ? i : m);
                int dd = AS == 2 ? s : (AL == 2 ? i : m);
                avt[((((long)b * 66 + hh + 1) * 66 + (ww + 1)) * 66 + (dd + 1)) * 192
                    + 64 * DST + c] = (__bf16)val;
            }
        }
    }
}

// ===========================================================================
// Kernel 3: final conv over cat(AVa,AVb,AVc) (Cin=192, bf16 padded AVt),
// fused bias + BN*ReLU*BN + residual(+q).  Slab per dh staged via TDM.
// ===========================================================================
__global__ void __launch_bounds__(256) out_conv_kernel(
    const __bf16* __restrict__ avt, const __bf16* __restrict__ wcp,
    const float* __restrict__ bc, const float* __restrict__ qo,
    float* __restrict__ y)
{
    __shared__ __attribute__((aligned(32))) __bf16 sx[3][34][192];  // [dw][d][cin] 38KB

    const int tid = threadIdx.x, blk = blockIdx.x;
    const int dt = blk & 1, w = (blk >> 1) & 63, h = (blk >> 7) & 63, b = blk >> 13;
    const int d0 = dt * 32;

    const int lane = tid & 31, wave = tid >> 5;
    const int mi = wave >> 1, ni = wave & 1;
    const int hf = lane >> 4, l15 = lane & 15;
    const int col = 16 * ni + l15;
    const long fbase = (long)mi * 512 + lane * 16;

    v8f acc = {};

    for (int dh = 0; dh < 3; ++dh) {
        __syncthreads();
        const long origin = ((((long)b * 66 + (h + dh)) * 66 + w) * 66 + d0) * 192;
#if HAVE_TDM
        if (tid == 0) {
            tdm_load((unsigned)(unsigned long long)(&sx[0][0][0]), avt + origin,
                     34 * 192, 3, 0, 66 * 192, 0, 34 * 192, 3, 1);
            __builtin_amdgcn_s_wait_tensorcnt(0);
        }
#else
        for (int ch = tid; ch < 2448; ch += 256) {       // 3 slabs * 816 x 16B
            int dw = ch / 816, rem = ch % 816;
            const u32x4* src = (const u32x4*)(avt + origin + (long)dw * (66 * 192));
            *((u32x4*)&sx[dw][0][0] + rem) = src[rem];
        }
#endif
        __syncthreads();

        for (int t9 = 0; t9 < 9; ++t9) {
            const int dw = t9 / 3, dd = t9 % 3;
            const int tap = dh * 9 + t9;                 // == dh*9 + dw*3 + dd
            for (int kc = 0; kc < 6; ++kc) {
                const long fo = fbase + (long)(tap * 6 + kc) * 2048;
                v16bf aF = *(const v16bf*)(wcp + fo);
                v16bf bX = *(const v16bf*)&sx[dw][col + dd][32 * kc + 16 * hf];
                acc = __builtin_amdgcn_wmma_f32_16x16x32_bf16(false, aF, false, bX,
                                                              (short)0, acc, false, false);
            }
        }
    }

#pragma unroll
    for (int r = 0; r < 8; ++r) {
        int   co   = 16 * mi + r + 8 * hf;
        int   d    = d0 + 16 * ni + l15;
        long  oidx = ((((long)b * 64 + co) * 64 + h) * 64 + w) * 64 + d;
        float val  = (acc[r] + bc[co]) * BN_SCALE;       // conv + BN(eval)
        val        = fmaxf(val, 0.0f) * BN_SCALE;        // ReLU + outer BN(eval)
        y[oidx]    = val + qo[oidx];                     // residual + q
    }
}

// ===========================================================================
// Orchestration.  Inputs: x, Wq, bq, Wk, bk, Wv, bv, Wc, bc (all f32).
// ===========================================================================
extern "C" void kernel_launch(void* const* d_in, const int* in_sizes, int n_in,
                              void* d_out, int out_size, void* d_ws, size_t ws_size,
                              hipStream_t stream) {
    const float* x  = (const float*)d_in[0];
    const float* Wq = (const float*)d_in[1];
    const float* bq = (const float*)d_in[2];
    const float* Wk = (const float*)d_in[3];
    const float* bk = (const float*)d_in[4];
    const float* Wv = (const float*)d_in[5];
    const float* bv = (const float*)d_in[6];
    const float* Wc = (const float*)d_in[7];
    const float* bc = (const float*)d_in[8];
    float*       y  = (float*)d_out;

    const long XT_N  = 2L * 66 * 66 * 66 * 64;    // 36,799,488
    const long AVT_N = 2L * 66 * 66 * 66 * 192;   // 110,398,464

    char* p = (char*)d_ws;
    __bf16* xt  = (__bf16*)p;  p += XT_N * 2;
    __bf16* avt = (__bf16*)p;  p += AVT_N * 2;
    __bf16* wqp = (__bf16*)p;  p += 110592L * 2;
    __bf16* wkp = (__bf16*)p;  p += 110592L * 2;
    __bf16* wvp = (__bf16*)p;  p += 110592L * 2;
    __bf16* wcp = (__bf16*)p;  p += 331776L * 2;
    float*  qo  = (float*)p;   p += NEL * 4;
    __bf16* qh  = (__bf16*)p;  p += NEL * 2;
    __bf16* kh  = (__bf16*)p;  p += NEL * 2;
    __bf16* vh  = (__bf16*)p;  p += NEL * 2;
    float*  AVa = (float*)p;   p += NEL * 4;
    float*  AVb = (float*)p;   p += NEL * 4;
    float*  AVc = (float*)p;   /* end */

    // 0) zero the padded bf16 tensors (xt + avt are adjacent)
    const long zchunks = (XT_N + AVT_N) / 8;      // 16B chunks
    zero_pad_kernel<<<(unsigned)((zchunks + 255) / 256), 256, 0, stream>>>((u32x4*)d_ws, zchunks);

    // 1) pack x (padded/transposed bf16) and weights (fragment-ordered bf16)
    xpack_kernel<<<131072, 256, 0, stream>>>(x, xt);
    wpack_kernel<<<432, 256, 0, stream>>>(Wq, wqp, 64, 2);
    wpack_kernel<<<432, 256, 0, stream>>>(Wk, wkp, 64, 2);
    wpack_kernel<<<432, 256, 0, stream>>>(Wv, wvp, 64, 2);
    wpack_kernel<<<1296, 256, 0, stream>>>(Wc, wcp, 192, 6);

    // 2) fused q/k/v projections (TDM-staged implicit GEMM)
    qkv_conv_kernel<<<16384, 256, 0, stream>>>(xt, wqp, wkp, wvp, bq, bk, bv,
                                               qo, qh, kh, vh);

    // 3) six axial passes: axis codes 0=H,1=W,2=D; (S,L,M) per reference permute
    axial_kernel<2, 0, 1, 0, 1><<<8192, 256, 0, stream>>>(qh, kh, vh, AVa, avt); // [B,C,D,H,W]
    axial_kernel<2, 1, 0, 0, 0><<<8192, 256, 0, stream>>>(qh, kh, vh, AVa, avt); // [B,C,D,W,H]
    axial_kernel<1, 0, 2, 1, 1><<<8192, 256, 0, stream>>>(qh, kh, vh, AVb, avt); // [B,C,W,H,D]
    axial_kernel<1, 2, 0, 1, 0><<<8192, 256, 0, stream>>>(qh, kh, vh, AVb, avt); // [B,C,W,D,H]
    axial_kernel<0, 1, 2, 2, 1><<<8192, 256, 0, stream>>>(qh, kh, vh, AVc, avt); // [B,C,H,W,D]
    axial_kernel<0, 2, 1, 2, 0><<<8192, 256, 0, stream>>>(qh, kh, vh, AVc, avt); // [B,C,H,D,W]

    // 4) output conv over cat (Cin=192) + BN/ReLU/BN + residual q
    out_conv_kernel<<<16384, 256, 0, stream>>>(avt, wcp, bc, qo, y);
}
